// YOLOv3_64690797412783
// MI455X (gfx1250) — matrix operand order repaired and checked by reading, not compiled, not run on hardware
//
#include <hip/hip_runtime.h>
#include <cstdint>
#include <cstddef>

// ---------------------------------------------------------------------------
// YOLOv3-608 (batch 2) forward for gfx1250 (MI455X).
// Convs = implicit GEMM with v_wmma_f32_16x16x32_f16 (f16 in, f32 acc).
// NHWC f16 activations; weights converted f32->f16 n-major per launch.
// B tile staged to LDS via GLOBAL_LOAD_ASYNC_TO_LDS_B128 when available.
// ---------------------------------------------------------------------------

typedef _Float16 half_t;
typedef __attribute__((ext_vector_type(16))) _Float16 v16h;
typedef __attribute__((ext_vector_type(8)))  _Float16 v8h;
typedef __attribute__((ext_vector_type(8)))  float    v8f;
typedef __attribute__((ext_vector_type(4)))  int      v4i;

#define TM 256          // M tile (pixels) per block (8 waves x 32 rows)
#define TN 64           // N tile (out channels) per block
#define LDSS 72         // LDS row stride in halves (144B: 16B aligned, bank-spread)

#if defined(__has_builtin)
#if __has_builtin(__builtin_amdgcn_global_load_async_to_lds_b128) && \
    __has_builtin(__builtin_amdgcn_s_wait_asynccnt)
#define USE_ASYNC_LDS 1
#endif
#endif

#ifdef USE_ASYNC_LDS
typedef __attribute__((address_space(1))) v4i GV4;   // global b128 payload
typedef __attribute__((address_space(3))) v4i LV4;   // LDS b128 payload
// flat->as3: low 32 bits of a flat LDS pointer are the LDS byte offset (ISA aperture rule)
__device__ __forceinline__ LV4* to_lds(void* p)        { return (LV4*)(uint32_t)(uintptr_t)p; }
__device__ __forceinline__ GV4* to_glob(const void* p) { return (GV4*)(uintptr_t)p; }
#endif

union Frag { v16h v; v8h h[2]; };

__device__ __forceinline__ float sigm(float x) { return 1.0f / (1.0f + expf(-x)); }

// ---------------------------------------------------------------------------
// Weight prep: f32 OIHW -> f16 [CoutPad][Kpad] (n-major), K=(ky*k+kx)*CinP+cin
// ---------------------------------------------------------------------------
__global__ void k_wprep(const float* __restrict__ w, int Cout, int Cin, int ksz,
                        int CinP, int Kpad, int CoutPad, half_t* __restrict__ w16) {
  size_t i = (size_t)blockIdx.x * 256 + threadIdx.x;
  size_t total = (size_t)Kpad * CoutPad;
  if (i >= total) return;
  int kg = (int)(i % Kpad);
  int n  = (int)(i / Kpad);
  int tap = kg / CinP;
  int cin = kg - tap * CinP;
  float v = 0.0f;
  if (n < Cout && cin < Cin && tap < ksz * ksz) {
    int ky = tap / ksz, kx = tap - ky * ksz;
    v = w[(((size_t)n * Cin + cin) * ksz + ky) * ksz + kx];
  }
  w16[i] = (half_t)v;
}

// ---------------------------------------------------------------------------
// Image: f32 NCHW (2,3,608,608) -> f16 NHWC with C padded to 4
// ---------------------------------------------------------------------------
__global__ void k_img(const float* __restrict__ img, half_t* __restrict__ dst) {
  int i = blockIdx.x * 256 + threadIdx.x;
  const int HW = 608 * 608;
  if (i >= 2 * HW) return;
  int n = i / HW;
  int pix = i - n * HW;
  for (int c = 0; c < 4; ++c) {
    float v = (c < 3) ? img[((size_t)n * 3 + c) * HW + pix] : 0.0f;
    dst[(size_t)i * 4 + c] = (half_t)v;
  }
}

// ---------------------------------------------------------------------------
// Implicit-GEMM conv + fused BN-scale/bias + leaky + residual + concat store.
// Block: 256 threads (8 wave32). Wave w owns rows [32w,32w+32) as two 16-row
// strips sharing B fragments -> 8 WMMAs per K-step per wave.
// ---------------------------------------------------------------------------
__global__ __launch_bounds__(256)
void k_conv_wmma(const half_t* __restrict__ in, int H, int W, int CsIn, int CinP,
                 const half_t* __restrict__ w16, int Kpad, int Kreal, int CoutPad,
                 int ksz, int stride,
                 const float* __restrict__ g, const float* __restrict__ bias,
                 const half_t* __restrict__ res, int CsRes,
                 half_t* __restrict__ out, int Hout, int Wout, int CsOut, int coOut,
                 int Cout, int leaky) {
  __shared__ __align__(16) half_t ldsA[TM * LDSS];
  __shared__ __align__(16) half_t ldsB[TN * LDSS];

  const int tid  = threadIdx.x;
  const int wave = tid >> 5;
  const int lane = tid & 31;
  const int hw   = Hout * Wout;
  const int M    = 2 * hw;                  // batch = 2
  const int pixBase = blockIdx.x * TM;
  const int nBase   = blockIdx.y * TN;
  const int pad = (ksz - 1) >> 1;
  const bool fast = ((CinP & 31) == 0);

  v8f acc[2][4] = {};

  // ---- hoisted im2col pixel decomposition (K-invariant) ----
  int pN[4], pOy[4], pOx[4];
  bool pV[4];
  if (fast) {
    int rbase = tid >> 2;                   // kloc = (tid&3)*8 fixed per thread
#pragma unroll
    for (int s = 0; s < 4; ++s) {
      int p = pixBase + rbase + 64 * s;
      pV[s] = p < M;
      int n = 0, oy = 0, ox = 0;
      if (pV[s]) { n = p / hw; int rp = p - n * hw; oy = rp / Wout; ox = rp - oy * Wout; }
      pN[s] = n; pOy[s] = oy; pOx[s] = ox;
    }
  } else {
    int p = pixBase + tid;                  // slow path: row == tid, 32 kk each
    pV[0] = p < M;
    int n = 0, oy = 0, ox = 0;
    if (pV[0]) { n = p / hw; int rp = p - n * hw; oy = rp / Wout; ox = rp - oy * Wout; }
    pN[0] = n; pOy[0] = oy; pOx[0] = ox;
  }

  for (int K0 = 0; K0 < Kpad; K0 += 32) {
    // -------- stage B (32 x 64, [n][kk]) from n-major weights --------
    {
      int nrow = tid >> 2;
      int kloc = (tid & 3) * 8;
      const half_t* gsrc = w16 + (size_t)(nBase + nrow) * Kpad + K0 + kloc;
      half_t* ldst = &ldsB[nrow * LDSS + kloc];
#ifdef USE_ASYNC_LDS
      __builtin_amdgcn_global_load_async_to_lds_b128(to_glob(gsrc), to_lds(ldst), 0, 0);
#else
      *reinterpret_cast<v8h*>(ldst) = *reinterpret_cast<const v8h*>(gsrc);
#endif
    }
    // -------- stage A (256 x 32) : im2col gather --------
    if (fast) {
      int kloc = (tid & 3) * 8;
      int kg   = K0 + kloc;
      int tap  = kg / CinP;                 // whole vec8 inside one tap (CinP%32==0)
      int cin0 = kg - tap * CinP;
      int ky = tap / ksz, kx = tap - ky * ksz;
      bool kv = kg < Kreal;
      int rbase = tid >> 2;
#pragma unroll
      for (int s = 0; s < 4; ++s) {
        v8h val = {};
        if (pV[s] && kv) {
          int iy = pOy[s] * stride + ky - pad;
          int ix = pOx[s] * stride + kx - pad;
          if (iy >= 0 && iy < H && ix >= 0 && ix < W)
            val = *reinterpret_cast<const v8h*>(in + ((size_t)(pN[s] * H + iy) * W + ix) * CsIn + cin0);
        }
        *reinterpret_cast<v8h*>(&ldsA[(rbase + 64 * s) * LDSS + kloc]) = val;
      }
    } else {                                // conv1 path (CinP==4): scalar gather
      for (int j = 0; j < 32; ++j) {
        int kg = K0 + j;
        half_t val = (half_t)0.0f;
        if (pV[0] && kg < Kreal) {
          int tap = kg / CinP, cin = kg - tap * CinP;
          int ky = tap / ksz, kx = tap - ky * ksz;
          int iy = pOy[0] * stride + ky - pad;
          int ix = pOx[0] * stride + kx - pad;
          if (iy >= 0 && iy < H && ix >= 0 && ix < W)
            val = in[((size_t)(pN[0] * H + iy) * W + ix) * CsIn + cin];
        }
        ldsA[tid * LDSS + j] = val;
      }
    }
    if (K0 + 32 < Kpad)                     // hint next weight slab (global_prefetch)
      __builtin_prefetch(w16 + (size_t)nBase * Kpad + K0 + 32, 0, 1);
#ifdef USE_ASYNC_LDS
    __builtin_amdgcn_s_wait_asynccnt(0);
#endif
    __syncthreads();

    // -------- fragments + WMMA (8 per wave per K-step) --------
    {
      int r16 = lane & 15;
      int kh  = lane >> 4;
      int arow0 = (wave << 5) + r16;        // strip 0
      int arow1 = arow0 + 16;               // strip 1
      Frag a0, a1;   // ISA 16-bit A 16x32: e<8 -> K=kh*8+e ; e>=8 -> K=16+kh*8+(e-8)
      a0.h[0] = *reinterpret_cast<const v8h*>(&ldsA[arow0 * LDSS + kh * 8]);
      a0.h[1] = *reinterpret_cast<const v8h*>(&ldsA[arow0 * LDSS + 16 + kh * 8]);
      a1.h[0] = *reinterpret_cast<const v8h*>(&ldsA[arow1 * LDSS + kh * 8]);
      a1.h[1] = *reinterpret_cast<const v8h*>(&ldsA[arow1 * LDSS + 16 + kh * 8]);
#pragma unroll
      for (int s = 0; s < 4; ++s) {
        Frag b;      // B 32x16: lane%16 = N col, e<8 -> K=kh*8+e ; e>=8 -> K=16+kh*8+(e-8)
        int n = s * 16 + r16;
        b.h[0] = *reinterpret_cast<const v8h*>(&ldsB[n * LDSS + kh * 8]);
        b.h[1] = *reinterpret_cast<const v8h*>(&ldsB[n * LDSS + 16 + kh * 8]);
        acc[0][s] = __builtin_amdgcn_wmma_f32_16x16x32_f16(
            false, a0.v, false, b.v, (short)0, acc[0][s], false, false);
        acc[1][s] = __builtin_amdgcn_wmma_f32_16x16x32_f16(
            false, a1.v, false, b.v, (short)0, acc[1][s], false, false);
      }
    }
    __syncthreads();
  }

  // -------- epilogue: C layout VGPR r -> M = r + (lane<16?0:8), N = lane%16 --------
  {
    int r16 = lane & 15;
    int mhi = (lane < 16) ? 0 : 8;
#pragma unroll
    for (int s = 0; s < 4; ++s) {
      int n = nBase + s * 16 + r16;
      if (n >= Cout) continue;
      float gg = g    ? g[n]    : 1.0f;
      float bb = bias ? bias[n] : 0.0f;
#pragma unroll
      for (int m = 0; m < 2; ++m) {
#pragma unroll
        for (int r = 0; r < 8; ++r) {
          int p = pixBase + (wave << 5) + m * 16 + r + mhi;
          if (p >= M) continue;
          float y = acc[m][s][r] * gg + bb;
          if (leaky) y = (y > 0.0f) ? y : 0.1f * y;
          if (res)   y += (float)res[(size_t)p * CsRes + n];
          out[(size_t)p * CsOut + coOut + n] = (half_t)y;
        }
      }
    }
  }
}

// ---------------------------------------------------------------------------
// 2x nearest upsample into (possibly concat) buffer
// ---------------------------------------------------------------------------
__global__ void k_up2(const half_t* __restrict__ in, int Hin, int C,
                      half_t* __restrict__ out, int CsOut, int coOut) {
  int Ho = 2 * Hin;
  size_t total = (size_t)2 * Ho * Ho * C;
  size_t i = (size_t)blockIdx.x * 256 + threadIdx.x;
  if (i >= total) return;
  int c = (int)(i % C);
  size_t p = i / C;
  int n  = (int)(p / ((size_t)Ho * Ho));
  int rp = (int)(p - (size_t)n * Ho * Ho);
  int oy = rp / Ho, ox = rp - oy * Ho;
  half_t v = in[(((size_t)n * Hin + (oy >> 1)) * Hin + (ox >> 1)) * C + c];
  out[p * CsOut + coOut + c] = v;
}

__global__ void k_copych(const half_t* __restrict__ in, int C, size_t npix,
                         half_t* __restrict__ out, int CsOut, int coOut) {
  size_t total = npix * C;
  size_t i = (size_t)blockIdx.x * 256 + threadIdx.x;
  if (i >= total) return;
  int c = (int)(i % C);
  size_t p = i / C;
  out[p * CsOut + coOut + c] = in[p * C + c];
}

// ---------------------------------------------------------------------------
// Decode: maps (f16, NHWC 18ch) -> boxes/scores (A = 22743 anchors/img)
// ---------------------------------------------------------------------------
#define NA 22743
__global__ void k_decode(const half_t* __restrict__ m0, const half_t* __restrict__ m1,
                         const half_t* __restrict__ m2,
                         float* __restrict__ boxes, float* __restrict__ scores) {
  int i = blockIdx.x * 256 + threadIdx.x;
  if (i >= 2 * NA) return;
  int img = i / NA;
  int a   = i - img * NA;
  const float anch[3][3][2] = {{{116,90},{156,198},{373,326}},
                               {{30,61},{62,45},{59,119}},
                               {{10,13},{16,30},{33,23}}};
  const half_t* mp; int gs, st, local, mi;
  if (a < 1083)      { mi = 0; gs = 19; st = 32; local = a;        mp = m0; }
  else if (a < 5415) { mi = 1; gs = 38; st = 16; local = a - 1083; mp = m1; }
  else               { mi = 2; gs = 76; st = 8;  local = a - 5415; mp = m2; }
  int anc = local % 3;
  int pix = local / 3;
  int gy = pix / gs, gx = pix - gy * gs;
  const half_t* q = mp + ((size_t)img * gs * gs + pix) * 18 + anc * 6;
  float tx = (float)q[0], ty = (float)q[1], tw = (float)q[2];
  float th = (float)q[3], to = (float)q[4], tc = (float)q[5];
  float obj = sigm(to), cls = sigm(tc);
  float cx = (gx + 0.5f) * st + (sigm(tx) - 0.5f) * st;
  float cy = (gy + 0.5f) * st + (sigm(ty) - 0.5f) * st;
  float w = anch[mi][anc][0] * expf(tw);
  float h = anch[mi][anc][1] * expf(th);
  float sc = obj * cls;
  if (!(obj >= 0.005f && cls > 0.05f)) sc = 0.0f;
  float* b = boxes + ((size_t)img * NA + a) * 4;
  b[0] = cx - 0.5f * w; b[1] = cy - 0.5f * h;
  b[2] = cx + 0.5f * w; b[3] = cy + 0.5f * h;
  scores[(size_t)img * NA + a] = sc;
}

// ---------------------------------------------------------------------------
// Per-image top-300 selection (destructive on a scratch copy)
// ---------------------------------------------------------------------------
__global__ void k_top300(const float* __restrict__ scores, float* __restrict__ scratch,
                         const float* __restrict__ boxes,
                         float* __restrict__ s300, float* __restrict__ b300) {
  int img = blockIdx.x, tid = threadIdx.x;
  const float* sc = scores + (size_t)img * NA;
  float* cp = scratch + (size_t)img * NA;
  for (int i = tid; i < NA; i += 256) cp[i] = sc[i];
  __syncthreads();
  __shared__ float sv[256];
  __shared__ int   si[256];
  for (int t = 0; t < 300; ++t) {
    float best = -1.0f; int bi = 0;
    for (int i = tid; i < NA; i += 256) { float v = cp[i]; if (v > best) { best = v; bi = i; } }
    sv[tid] = best; si[tid] = bi;
    __syncthreads();
    for (int off = 128; off > 0; off >>= 1) {
      if (tid < off && sv[tid + off] > sv[tid]) { sv[tid] = sv[tid + off]; si[tid] = si[tid + off]; }
      __syncthreads();
    }
    if (tid == 0) {
      int k = si[0];
      s300[img * 300 + t] = sc[k];
      for (int c = 0; c < 4; ++c)
        b300[((size_t)img * 300 + t) * 4 + c] = boxes[((size_t)img * NA + k) * 4 + c];
      cp[k] = -1.0f;
    }
    __syncthreads();
  }
}

// ---------------------------------------------------------------------------
// NMS (sequential suppression) + top-100 emit into d_out
// d_out: dets (2,100,5) then classes (2,100) all-zero (NUM_CLASSES==1)
// ---------------------------------------------------------------------------
__global__ void k_nms(const float* __restrict__ b300, const float* __restrict__ s300,
                      float* __restrict__ dout) {
  int img = blockIdx.x, tid = threadIdx.x;
  __shared__ float bx[300][4];
  __shared__ float ss[300];
  __shared__ int   keep[300];
  __shared__ float sk[300];
  __shared__ int   supFlag;
  __shared__ float rv[256];
  __shared__ int   ri[256];
  for (int i = tid; i < 300; i += 256) {
    for (int c = 0; c < 4; ++c) bx[i][c] = b300[((size_t)img * 300 + i) * 4 + c];
    ss[i] = s300[img * 300 + i];
    keep[i] = 0;
  }
  __syncthreads();
  for (int i = 0; i < 300; ++i) {
    if (tid == 0) supFlag = 0;
    __syncthreads();
    float a0 = bx[i][0], a1 = bx[i][1], a2 = bx[i][2], a3 = bx[i][3];
    float aa = (a2 - a0) * (a3 - a1);
    int local = 0;
    for (int j = tid; j < i; j += 256) {
      if (keep[j]) {
        float lt0 = fmaxf(a0, bx[j][0]), lt1 = fmaxf(a1, bx[j][1]);
        float rb0 = fminf(a2, bx[j][2]), rb1 = fminf(a3, bx[j][3]);
        float w = fmaxf(rb0 - lt0, 0.0f), h = fmaxf(rb1 - lt1, 0.0f);
        float inter = w * h;
        float ab = (bx[j][2] - bx[j][0]) * (bx[j][3] - bx[j][1]);
        float iou = inter / fmaxf(aa + ab - inter, 1e-9f);
        if (iou > 0.45f) local = 1;
      }
    }
    if (local) supFlag = 1;                 // benign race: same value
    __syncthreads();
    if (tid == 0) keep[i] = (!supFlag) && (ss[i] > 0.0f);
    __syncthreads();
  }
  for (int i = tid; i < 300; i += 256) sk[i] = keep[i] ? ss[i] : 0.0f;
  __syncthreads();
  for (int t = 0; t < 100; ++t) {
    float best = -2.0f; int bi = 0;
    for (int i = tid; i < 300; i += 256) { float v = sk[i]; if (v > best) { best = v; bi = i; } }
    rv[tid] = best; ri[tid] = bi;
    __syncthreads();
    for (int off = 128; off > 0; off >>= 1) {
      if (tid < off && rv[tid + off] > rv[tid]) { rv[tid] = rv[tid + off]; ri[tid] = ri[tid + off]; }
      __syncthreads();
    }
    if (tid == 0) {
      int k = ri[0];
      float* d = dout + ((size_t)img * 100 + t) * 5;
      d[0] = bx[k][0]; d[1] = bx[k][1]; d[2] = bx[k][2]; d[3] = bx[k][3];
      d[4] = rv[0];
      sk[k] = -1.0f;
    }
    __syncthreads();
  }
  for (int i = tid; i < 100; i += 256) dout[1000 + img * 100 + i] = 0.0f;  // classes
}

// ---------------------------------------------------------------------------
// Host orchestration
// ---------------------------------------------------------------------------
struct CU { const float *b, *g, *w; };
struct PR { const float *b, *w; };

extern "C" void kernel_launch(void* const* d_in, const int* in_sizes, int n_in,
                              void* d_out, int out_size, void* d_ws, size_t ws_size,
                              hipStream_t stream) {
  (void)in_sizes; (void)n_in; (void)out_size;
  const float* imgs = (const float*)d_in[0];

  // ---- param mapping: JAX dict flatten order (sorted keys) ----
  // params: backbone < head < neck ; backbone: blocks < conv1 ; block: down < res
  // conv-unit dict {'w','g','b'} flattens b,g,w ; pred {'w','b'} flattens b,w
  int t = 1;
  auto nextCU = [&]() { CU c{(const float*)d_in[t], (const float*)d_in[t + 1], (const float*)d_in[t + 2]}; t += 3; return c; };
  int L[5] = {1, 2, 8, 8, 4};
  CU down[5], resA[23], resB[23];
  int ri = 0;
  for (int i = 0; i < 5; ++i) {
    down[i] = nextCU();
    for (int j = 0; j < L[i]; ++j) { resA[ri] = nextCU(); resB[ri] = nextCU(); ++ri; }
  }
  CU cv1 = nextCU();
  CU bridge[3]; for (int i = 0; i < 3; ++i) bridge[i] = nextCU();
  PR pred[3];   for (int i = 0; i < 3; ++i) { pred[i] = {(const float*)d_in[t], (const float*)d_in[t + 1]}; t += 2; }
  CU c1 = nextCU(), c2 = nextCU();
  CU d1[5], d2[5], d3[5];
  for (int i = 0; i < 5; ++i) d1[i] = nextCU();
  for (int i = 0; i < 5; ++i) d2[i] = nextCU();
  for (int i = 0; i < 5; ++i) d3[i] = nextCU();

  // ---- workspace bump allocator ----
  char* base = (char*)d_ws;
  size_t off = 0;
  auto alloc  = [&](size_t bytes) -> char* { char* p = base + off; off = (off + bytes + 255) & ~(size_t)255; return p; };
  auto allocH = [&](size_t elems) -> half_t* { return (half_t*)alloc(elems * 2); };
  auto E = [](int H, int C) { return (size_t)2 * H * H * C; };

  half_t* IMG   = allocH(E(608, 4));
  half_t* B608  = allocH(E(608, 32));
  half_t* P304a = allocH(E(304, 64));  half_t* P304b = allocH(E(304, 64));  half_t* T304 = allocH(E(304, 32));
  half_t* P152a = allocH(E(152, 128)); half_t* P152b = allocH(E(152, 128)); half_t* T152 = allocH(E(152, 64));
  half_t* P76a  = allocH(E(76, 256));  half_t* P76b  = allocH(E(76, 256));  half_t* T76  = allocH(E(76, 128));
  half_t* P38a  = allocH(E(38, 512));  half_t* P38b  = allocH(E(38, 512));  half_t* T38  = allocH(E(38, 256));
  half_t* P19a  = allocH(E(19, 1024)); half_t* P19b  = allocH(E(19, 1024)); half_t* T19  = allocH(E(19, 512));
  half_t* X1 = allocH(E(76, 256)); half_t* X2 = allocH(E(38, 512)); half_t* X3 = allocH(E(19, 1024));
  half_t* N19a = allocH(E(19, 512)); half_t* N19b = allocH(E(19, 1024)); half_t* Y3 = allocH(E(19, 512));
  half_t* C38  = allocH(E(38, 768)); half_t* N38a = allocH(E(38, 256)); half_t* N38b = allocH(E(38, 512)); half_t* Y2 = allocH(E(38, 256));
  half_t* C76  = allocH(E(76, 384)); half_t* N76a = allocH(E(76, 128)); half_t* N76b = allocH(E(76, 256)); half_t* Y1 = allocH(E(76, 128));
  half_t* Hd19 = allocH(E(19, 1024)); half_t* Hd38 = allocH(E(38, 512)); half_t* Hd76 = allocH(E(76, 256));
  half_t* Mp0 = allocH(E(19, 18)); half_t* Mp1 = allocH(E(38, 18)); half_t* Mp2 = allocH(E(76, 18));
  half_t* WSw = allocH((size_t)4608 * 1024);      // largest Kpad*CoutPad (512->1024 k3)
  float* BOX  = (float*)alloc((size_t)2 * NA * 4 * 4);
  float* SCRS = (float*)alloc((size_t)2 * NA * 4);
  float* SCT  = (float*)alloc((size_t)2 * NA * 4);
  float* S300 = (float*)alloc((size_t)2 * 300 * 4);
  float* B300 = (float*)alloc((size_t)2 * 300 * 4 * 4);
  if (off > ws_size) return;                       // ws too small: bail deterministically

  // ---- conv helpers ----
  auto conv = [&](const half_t* in, int H, int CsIn, int Cin, int CinP,
                  const float* w, const float* g, const float* b,
                  int Cout, int ksz, int stride,
                  const half_t* res, int CsRes,
                  half_t* outp, int CsOut, int coOut, int leaky) {
    int Kreal = CinP * ksz * ksz;
    int Kpad  = (Kreal + 31) & ~31;
    int CoutPad = (Cout + 63) & ~63;
    size_t tot = (size_t)Kpad * CoutPad;
    k_wprep<<<dim3((unsigned)((tot + 255) / 256)), dim3(256), 0, stream>>>(
        w, Cout, Cin, ksz, CinP, Kpad, CoutPad, WSw);
    int Hout = (stride == 2) ? H / 2 : H;
    int Mtot = 2 * Hout * Hout;
    dim3 grid((Mtot + TM - 1) / TM, CoutPad / TN);
    k_conv_wmma<<<grid, dim3(256), 0, stream>>>(
        in, H, H, CsIn, CinP, WSw, Kpad, Kreal, CoutPad, ksz, stride,
        g, b, res, CsRes, outp, Hout, Hout, CsOut, coOut, Cout, leaky);
  };
  auto convU = [&](const half_t* in, int H, int CsIn, const CU& cu, int Cin, int Cout,
                   int ksz, int stride, const half_t* res, half_t* outp, int CsOut, int coOut) {
    conv(in, H, CsIn, Cin, Cin, cu.w, cu.g, cu.b, Cout, ksz, stride, res, Cout, outp, CsOut, coOut, 1);
  };
  auto up2 = [&](const half_t* in, int Hin, int C, half_t* outp, int CsOut, int co) {
    size_t total = (size_t)2 * (2 * Hin) * (2 * Hin) * C;
    k_up2<<<dim3((unsigned)((total + 255) / 256)), dim3(256), 0, stream>>>(in, Hin, C, outp, CsOut, co);
  };
  auto copych = [&](const half_t* in, int C, int Hp, half_t* outp, int CsOut, int co) {
    size_t npix = (size_t)2 * Hp * Hp;
    size_t total = npix * C;
    k_copych<<<dim3((unsigned)((total + 255) / 256)), dim3(256), 0, stream>>>(in, C, npix, outp, CsOut, co);
  };

  // ---- backbone ----
  k_img<<<dim3((2 * 608 * 608 + 255) / 256), dim3(256), 0, stream>>>(imgs, IMG);
  conv(IMG, 608, 4, 3, 4, cv1.w, cv1.g, cv1.b, 32, 3, 1, nullptr, 0, B608, 32, 0, 1);

  half_t* PA[5] = {P304a, P152a, P76a, P38a, P19a};
  half_t* PB[5] = {P304b, P152b, P76b, P38b, P19b};
  half_t* TT[5] = {T304, T152, T76, T38, T19};
  half_t* XF[5] = {nullptr, nullptr, X1, X2, X3};
  int CC[5] = {64, 128, 256, 512, 1024};
  int HB[5] = {304, 152, 76, 38, 19};
  half_t* cur = B608;
  int Hcur = 608;
  ri = 0;
  for (int b = 0; b < 5; ++b) {
    int Cin = (b == 0) ? 32 : CC[b - 1];
    convU(cur, Hcur, Cin, down[b], Cin, CC[b], 3, 2, nullptr, PA[b], CC[b], 0);
    half_t* x = PA[b];
    for (int j = 0; j < L[b]; ++j) {
      half_t* dst = (j == L[b] - 1 && XF[b]) ? XF[b] : ((x == PA[b]) ? PB[b] : PA[b]);
      convU(x, HB[b], CC[b], resA[ri], CC[b], CC[b] / 2, 1, 1, nullptr, TT[b], CC[b] / 2, 0);
      convU(TT[b], HB[b], CC[b] / 2, resB[ri], CC[b] / 2, CC[b], 3, 1, x, dst, CC[b], 0);
      x = dst; ++ri;
    }
    cur = x; Hcur = HB[b];
  }
  // cur == X3 (1024@19), X2 (512@38), X1 (256@76)

  // ---- neck ----
  convU(X3,   19, 1024, d1[0], 1024, 512,  1, 1, nullptr, N19a, 512,  0);
  convU(N19a, 19, 512,  d1[1], 512,  1024, 3, 1, nullptr, N19b, 1024, 0);
  convU(N19b, 19, 1024, d1[2], 1024, 512,  1, 1, nullptr, N19a, 512,  0);
  convU(N19a, 19, 512,  d1[3], 512,  1024, 3, 1, nullptr, N19b, 1024, 0);
  convU(N19b, 19, 1024, d1[4], 1024, 512,  1, 1, nullptr, Y3,   512,  0);
  convU(Y3,   19, 512,  c1,    512,  256,  1, 1, nullptr, N19a, 256,  0);   // reuse N19a as 256ch
  up2(N19a, 19, 256, C38, 768, 0);
  copych(X2, 512, 38, C38, 768, 256);
  convU(C38,  38, 768,  d2[0], 768,  256, 1, 1, nullptr, N38a, 256, 0);
  convU(N38a, 38, 256,  d2[1], 256,  512, 3, 1, nullptr, N38b, 512, 0);
  convU(N38b, 38, 512,  d2[2], 512,  256, 1, 1, nullptr, N38a, 256, 0);
  convU(N38a, 38, 256,  d2[3], 256,  512, 3, 1, nullptr, N38b, 512, 0);
  convU(N38b, 38, 512,  d2[4], 512,  256, 1, 1, nullptr, Y2,   256, 0);
  convU(Y2,   38, 256,  c2,    256,  128, 1, 1, nullptr, N38a, 128, 0);     // reuse N38a as 128ch
  up2(N38a, 38, 128, C76, 384, 0);
  copych(X1, 256, 76, C76, 384, 128);
  convU(C76,  76, 384,  d3[0], 384,  128, 1, 1, nullptr, N76a, 128, 0);
  convU(N76a, 76, 128,  d3[1], 128,  256, 3, 1, nullptr, N76b, 256, 0);
  convU(N76b, 76, 256,  d3[2], 256,  128, 1, 1, nullptr, N76a, 128, 0);
  convU(N76a, 76, 128,  d3[3], 128,  256, 3, 1, nullptr, N76b, 256, 0);
  convU(N76b, 76, 256,  d3[4], 256,  128, 1, 1, nullptr, Y1,   128, 0);

  // ---- head ----
  convU(Y3, 19, 512, bridge[0], 512, 1024, 3, 1, nullptr, Hd19, 1024, 0);
  conv(Hd19, 19, 1024, 1024, 1024, pred[0].w, nullptr, pred[0].b, 18, 1, 1, nullptr, 0, Mp0, 18, 0, 0);
  convU(Y2, 38, 256, bridge[1], 256, 512, 3, 1, nullptr, Hd38, 512, 0);
  conv(Hd38, 38, 512, 512, 512, pred[1].w, nullptr, pred[1].b, 18, 1, 1, nullptr, 0, Mp1, 18, 0, 0);
  convU(Y1, 76, 128, bridge[2], 128, 256, 3, 1, nullptr, Hd76, 256, 0);
  conv(Hd76, 76, 256, 256, 256, pred[2].w, nullptr, pred[2].b, 18, 1, 1, nullptr, 0, Mp2, 18, 0, 0);

  // ---- postprocess ----
  k_decode<<<dim3((2 * NA + 255) / 256), dim3(256), 0, stream>>>(Mp0, Mp1, Mp2, BOX, SCRS);
  k_top300<<<dim3(2), dim3(256), 0, stream>>>(SCRS, SCT, BOX, S300, B300);
  k_nms<<<dim3(2), dim3(256), 0, stream>>>(B300, S300, (float*)d_out);
}
